// AttentionBlock_85529978733036
// MI455X (gfx1250) — matrix operand order repaired and checked by reading, not compiled, not run on hardware
//
#include <hip/hip_runtime.h>

// ---- problem constants (match reference) ----
#define BB   8
#define SS   4096
#define EE   512
#define NHH  8
#define DHH  64
#define HIDD 2048

typedef __attribute__((ext_vector_type(16))) __bf16 v16bf;
typedef __attribute__((ext_vector_type(8)))  __bf16 v8bf;
typedef __attribute__((ext_vector_type(4)))  __bf16 v4bf;
typedef __attribute__((ext_vector_type(8)))  float  v8f;

#define EPI_F32  0
#define EPI_BF16 1
#define EPI_ELU1 2
#define EPI_RELU 3

// ---------------- f32 -> bf16 convert ----------------
__global__ __launch_bounds__(256)
void cvt_f32_bf16_kernel(const float* __restrict__ in, __bf16* __restrict__ out, int n4) {
    int i = blockIdx.x * 256 + threadIdx.x;
    if (i < n4) {
        float4 v = ((const float4*)in)[i];
        v4bf o;
        o[0] = (__bf16)v.x; o[1] = (__bf16)v.y; o[2] = (__bf16)v.z; o[3] = (__bf16)v.w;
        ((v4bf*)out)[i] = o;
    }
}

// ---------------- bf16 WMMA GEMM: C = A(MxK) * B(KxN) + bias, fused epilogue ----
// 256 threads = 8 waves (4 row-groups x 2 col-groups). Block tile 128x128, BK=64.
// Each wave: 32x64 sub-tile = 2 A-frags x 4 B-frags = 8 WMMA per 32-K-step.
// Double-buffered LDS, one barrier per K-tile; global prefetch overlaps compute.
template<int EPI>
__global__ __launch_bounds__(256)
void gemm_bf16_kernel(const __bf16* __restrict__ A, const __bf16* __restrict__ Bw,
                      const float* __restrict__ bias,
                      float* __restrict__ outF, __bf16* __restrict__ outH,
                      int M, int N, int K) {
    constexpr int BM = 128, BN = 128, BK = 64;
    constexpr int LSTR = BK + 8;  // 72 bf16 -> 144B rows (16B aligned)
    __shared__ __align__(16) __bf16 As[2][BM * LSTR];
    __shared__ __align__(16) __bf16 BsT[2][BN * LSTR];

    const int t    = threadIdx.x;
    const int lane = t & 31;
    const int wave = t >> 5;
    const int wr   = wave >> 1;   // 0..3 -> 32-row group
    const int wc   = wave & 1;    // 0..1 -> 64-col group
    const int hb   = lane >> 4;   // half of wave
    const int l16  = lane & 15;
    const int m0   = blockIdx.y * BM;
    const int n0   = blockIdx.x * BN;

    v8f acc[2][4];
#pragma unroll
    for (int i = 0; i < 2; ++i)
#pragma unroll
        for (int j = 0; j < 4; ++j)
#pragma unroll
            for (int e = 0; e < 8; ++e) acc[i][j][e] = 0.f;

    v8bf aReg[4], bReg[4];

    // A tile: 128 rows x 64 k  (1024 v8bf segs). B tile: 64 k x 128 n.
    auto globalLoad = [&](int k0) {
#pragma unroll
        for (int i = 0; i < 4; ++i) {
            const int s  = t + 256 * i;
            const int ar = s >> 3, ac = (s & 7) * 8;
            aReg[i] = *(const v8bf*)(A + (size_t)(m0 + ar) * K + k0 + ac);
            const int br = s >> 4, bc = (s & 15) * 8;
            bReg[i] = *(const v8bf*)(Bw + (size_t)(k0 + br) * N + n0 + bc);
        }
    };
    auto ldsStore = [&](int buf) {
#pragma unroll
        for (int i = 0; i < 4; ++i) {
            const int s  = t + 256 * i;
            const int ar = s >> 3, ac = (s & 7) * 8;
            *(v8bf*)&As[buf][ar * LSTR + ac] = aReg[i];
            const int br = s >> 4, bc = (s & 15) * 8;
#pragma unroll
            for (int j = 0; j < 8; ++j)
                BsT[buf][(bc + j) * LSTR + br] = bReg[i][j];   // transpose: BsT[n][k]
        }
    };
    auto compute = [&](int buf) {
#pragma unroll
        for (int kk = 0; kk < BK; kk += 32) {
            v16bf a[2], b[4];
#pragma unroll
            for (int i = 0; i < 2; ++i) {
                // A frag: lane l16 = row; K = hb*16 .. +15 (contiguous)
                const __bf16* ap = &As[buf][(wr * 32 + i * 16 + l16) * LSTR + kk];
                v8bf lo = *(const v8bf*)(ap + hb * 16);
                v8bf hi = *(const v8bf*)(ap + hb * 16 + 8);
#pragma unroll
                for (int j = 0; j < 8; ++j) { a[i][j] = lo[j]; a[i][8 + j] = hi[j]; }
            }
#pragma unroll
            for (int ns = 0; ns < 4; ++ns) {
                // B frag: lane l16 = col; K = hb*16 .. +15 (contiguous in BsT)
                const __bf16* bp = &BsT[buf][(wc * 64 + ns * 16 + l16) * LSTR + kk + hb * 16];
                v8bf lo = *(const v8bf*)bp;
                v8bf hi = *(const v8bf*)(bp + 8);
#pragma unroll
                for (int j = 0; j < 8; ++j) { b[ns][j] = lo[j]; b[ns][8 + j] = hi[j]; }
            }
#pragma unroll
            for (int i = 0; i < 2; ++i)
#pragma unroll
                for (int ns = 0; ns < 4; ++ns)
                    acc[i][ns] = __builtin_amdgcn_wmma_f32_16x16x32_bf16(
                        false, a[i], false, b[ns], (short)0, acc[i][ns], false, false);
        }
    };

    globalLoad(0);
    ldsStore(0);
    __syncthreads();
    const int nt = K / BK;
    for (int it = 0; it < nt; ++it) {
        if (it + 1 < nt) globalLoad((it + 1) * BK);
        compute(it & 1);
        if (it + 1 < nt) ldsStore((it + 1) & 1);
        __syncthreads();
    }

    // Epilogue. D layout: VGPR r -> row (i*16 + hb*8 + r), col (ns*16 + l16)
#pragma unroll
    for (int i = 0; i < 2; ++i) {
#pragma unroll
        for (int ns = 0; ns < 4; ++ns) {
            const int col = n0 + wc * 64 + ns * 16 + l16;
            const float bv = bias[col];
#pragma unroll
            for (int r = 0; r < 8; ++r) {
                const int row = m0 + wr * 32 + i * 16 + hb * 8 + r;
                float v = acc[i][ns][r] + bv;
                size_t idx = (size_t)row * N + col;
                if (EPI == EPI_F32)  outF[idx] = v;
                if (EPI == EPI_BF16) outH[idx] = (__bf16)v;
                if (EPI == EPI_ELU1) outH[idx] = (__bf16)(v >= 0.f ? v + 1.f : __expf(v));
                if (EPI == EPI_RELU) outH[idx] = (__bf16)fmaxf(v, 0.f);
            }
        }
    }
}

// ---------------- KV[b,h,m,d] = sum_s V[s,m] K[s,d];  Ksum[b,h,d] = sum_s K[s,d] ----
// grid = (32 s-chunks, B*NH); float atomics into zeroed KV/Ksum.
__global__ __launch_bounds__(256)
void kv_ksum_kernel(const __bf16* __restrict__ Kh, const __bf16* __restrict__ Vh,
                    float* __restrict__ KV, float* __restrict__ Ksum) {
    const int bh = blockIdx.y;
    const int b = bh >> 3, h = bh & 7;
    const int t = threadIdx.x;
    __shared__ __align__(16) __bf16 Ks[8][64];
    __shared__ __align__(16) __bf16 Vs[8][64];

    const int schunk  = SS / 32;
    const int s_begin = blockIdx.x * schunk;
    const size_t base = (size_t)b * SS * EE + (size_t)h * DHH;

    const int m  = t >> 2;
    const int db = (t & 3) * 16;

    float acc[16];
#pragma unroll
    for (int j = 0; j < 16; ++j) acc[j] = 0.f;
    float ks = 0.f;

    for (int s0 = s_begin; s0 < s_begin + schunk; s0 += 8) {
        __syncthreads();
        if (t < 128) {
            const int row = t >> 4, colb = (t & 15) * 4;
            *(v4bf*)&Ks[row][colb] =
                *(const v4bf*)(Kh + base + (size_t)(s0 + row) * EE + colb);
        } else {
            const int t2 = t - 128;
            const int row = t2 >> 4, colb = (t2 & 15) * 4;
            *(v4bf*)&Vs[row][colb] =
                *(const v4bf*)(Vh + base + (size_t)(s0 + row) * EE + colb);
        }
        __syncthreads();
#pragma unroll
        for (int ss = 0; ss < 8; ++ss) {
            float vm = (float)Vs[ss][m];
            v8bf k0 = *(const v8bf*)&Ks[ss][db];
            v8bf k1 = *(const v8bf*)&Ks[ss][db + 8];
#pragma unroll
            for (int j = 0; j < 8; ++j) {
                acc[j]     += vm * (float)k0[j];
                acc[8 + j] += vm * (float)k1[j];
            }
            if (t < 64) ks += (float)Ks[ss][t];
        }
    }
    float* dst = KV + ((size_t)bh * 64 + m) * 64 + db;
#pragma unroll
    for (int j = 0; j < 16; ++j) atomicAdd(dst + j, acc[j]);
    if (t < 64) atomicAdd(Ksum + bh * 64 + t, ks);
}

// ---------------- attn[b,s,h,m] = z * sum_d Q[d]*KV[m,d],  z = 1/(Q.Ksum + eps) ----
__global__ __launch_bounds__(256)
void attn_combine_kernel(const __bf16* __restrict__ Qh, const float* __restrict__ KV,
                         const float* __restrict__ Ksum, __bf16* __restrict__ Ah) {
    const int bh = blockIdx.y;
    const int b = bh >> 3, h = bh & 7;
    const int s0 = blockIdx.x * 32;
    const int t = threadIdx.x;
    __shared__ __align__(16) float KVs[64 * 64];
    __shared__ float Ksums[64];
    __shared__ __align__(16) __bf16 Qs[32 * 64];

    {
        const float4* src = (const float4*)(KV + (size_t)bh * 4096);
        float4* dst = (float4*)KVs;
        for (int i = t; i < 1024; i += 256) dst[i] = src[i];
    }
    if (t < 64) Ksums[t] = Ksum[bh * 64 + t];
    {
        const int row = t >> 3, colb = (t & 7) * 8;
        *(v8bf*)&Qs[row * 64 + colb] =
            *(const v8bf*)(Qh + (size_t)b * SS * EE + (size_t)(s0 + row) * EE + h * DHH + colb);
    }
    __syncthreads();

    const int r = t >> 3, mb = (t & 7) * 8;
    float qf[64];
#pragma unroll
    for (int d8 = 0; d8 < 8; ++d8) {
        v8bf q8 = *(const v8bf*)&Qs[r * 64 + d8 * 8];
#pragma unroll
        for (int j = 0; j < 8; ++j) qf[d8 * 8 + j] = (float)q8[j];
    }
    float z = 0.f;
#pragma unroll
    for (int d = 0; d < 64; ++d) z += qf[d] * Ksums[d];
    z = 1.f / (z + 1e-6f);

    __bf16* outp = Ah + (size_t)b * SS * EE + (size_t)(s0 + r) * EE + h * DHH + mb;
#pragma unroll
    for (int m = 0; m < 8; ++m) {
        const float* kv = &KVs[(mb + m) * 64];
        float a = 0.f;
#pragma unroll
        for (int d = 0; d < 64; ++d) a += qf[d] * kv[d];
        outp[m] = (__bf16)(a * z);
    }
}

// ---------------- residual add + LayerNorm (block per row, E=512) ----------------
template<bool WRITE_BF16>
__global__ __launch_bounds__(256)
void add_ln_kernel(const float* __restrict__ xin, const float* __restrict__ yin,
                   const float* __restrict__ g, const float* __restrict__ be,
                   float* __restrict__ outF, __bf16* __restrict__ outH) {
    const int row = blockIdx.x;
    const int t = threadIdx.x;
    const size_t base = (size_t)row * EE;
    float v0 = xin[base + t] + yin[base + t];
    float v1 = xin[base + t + 256] + yin[base + t + 256];

    __shared__ float2 red[256];
    red[t] = make_float2(v0 + v1, v0 * v0 + v1 * v1);
    __syncthreads();
    for (int off = 128; off > 0; off >>= 1) {
        if (t < off) {
            red[t].x += red[t + off].x;
            red[t].y += red[t + off].y;
        }
        __syncthreads();
    }
    const float mean = red[0].x * (1.f / EE);
    const float var  = red[0].y * (1.f / EE) - mean * mean;
    const float rstd = rsqrtf(var + 1e-5f);
    float o0 = (v0 - mean) * rstd * g[t] + be[t];
    float o1 = (v1 - mean) * rstd * g[t + 256] + be[t + 256];
    outF[base + t] = o0;
    outF[base + t + 256] = o1;
    if (WRITE_BF16) {
        outH[base + t] = (__bf16)o0;
        outH[base + t + 256] = (__bf16)o1;
    }
}

// ---------------- driver ----------------
extern "C" void kernel_launch(void* const* d_in, const int* in_sizes, int n_in,
                              void* d_out, int out_size, void* d_ws, size_t ws_size,
                              hipStream_t stream) {
    const float* x  = (const float*)d_in[0];
    const float* Wq = (const float*)d_in[1];
    const float* bq = (const float*)d_in[2];
    const float* Wk = (const float*)d_in[3];
    const float* bk = (const float*)d_in[4];
    const float* Wv = (const float*)d_in[5];
    const float* bv = (const float*)d_in[6];
    const float* Wo = (const float*)d_in[7];
    const float* bo = (const float*)d_in[8];
    const float* g1 = (const float*)d_in[9];
    const float* be1= (const float*)d_in[10];
    const float* g2 = (const float*)d_in[11];
    const float* be2= (const float*)d_in[12];
    const float* W1 = (const float*)d_in[13];
    const float* b1 = (const float*)d_in[14];
    const float* W2 = (const float*)d_in[15];
    const float* b2 = (const float*)d_in[16];

    const size_t MSE = (size_t)BB * SS * EE;
    const size_t MSH = (size_t)BB * SS * HIDD;

    char* ws = (char*)d_ws;
    size_t off = 0;
    auto wsalloc = [&](size_t bytes) -> void* {
        void* p = ws + off;
        off += (bytes + 255) & ~(size_t)255;
        return p;
    };

    __bf16* xh  = (__bf16*)wsalloc(MSE * 2);           // reused as Ah after QKV
    __bf16* Qh  = (__bf16*)wsalloc(MSE * 2);           // Qh+Kh region reused as t0
    __bf16* Kh  = (__bf16*)wsalloc(MSE * 2);
    __bf16* Vh  = (__bf16*)wsalloc(MSE * 2);           // reused as x1h after KV
    float*  x1  = (float*)wsalloc(MSE * 4);
    __bf16* Hh  = (__bf16*)wsalloc(MSH * 2);
    float*  KV  = (float*)wsalloc((size_t)BB * NHH * DHH * DHH * 4);
    float*  Ksum= (float*)wsalloc((size_t)BB * NHH * DHH * 4);
    __bf16* Wqh = (__bf16*)wsalloc((size_t)EE * EE * 2);
    __bf16* Wkh = (__bf16*)wsalloc((size_t)EE * EE * 2);
    __bf16* Wvh = (__bf16*)wsalloc((size_t)EE * EE * 2);
    __bf16* Woh = (__bf16*)wsalloc((size_t)EE * EE * 2);
    __bf16* W1h = (__bf16*)wsalloc((size_t)EE * HIDD * 2);
    __bf16* W2h = (__bf16*)wsalloc((size_t)HIDD * EE * 2);

    float*  t0  = (float*)Qh;   // 64MB spanning Qh+Kh (Q/K dead when written)
    __bf16* Ah  = xh;           // xh dead after QKV GEMMs
    __bf16* x1h = Vh;           // V dead after kv_ksum

    const dim3 blk(256);
    const int wE = (int)((size_t)EE * EE / 4);
    const int wH = (int)((size_t)EE * HIDD / 4);

    // converts
    cvt_f32_bf16_kernel<<<(unsigned)((MSE / 4 + 255) / 256), blk, 0, stream>>>(x, xh, (int)(MSE / 4));
    cvt_f32_bf16_kernel<<<(wE + 255) / 256, blk, 0, stream>>>(Wq, Wqh, wE);
    cvt_f32_bf16_kernel<<<(wE + 255) / 256, blk, 0, stream>>>(Wk, Wkh, wE);
    cvt_f32_bf16_kernel<<<(wE + 255) / 256, blk, 0, stream>>>(Wv, Wvh, wE);
    cvt_f32_bf16_kernel<<<(wE + 255) / 256, blk, 0, stream>>>(Wo, Woh, wE);
    cvt_f32_bf16_kernel<<<(wH + 255) / 256, blk, 0, stream>>>(W1, W1h, wH);
    cvt_f32_bf16_kernel<<<(wH + 255) / 256, blk, 0, stream>>>(W2, W2h, wH);

    const int M = BB * SS;
    const dim3 gE(EE / 128, M / 128);      // (4, 256)
    const dim3 gH(HIDD / 128, M / 128);    // (16, 256)

    // Q/K with elu+1 feature map, V plain
    gemm_bf16_kernel<EPI_ELU1><<<gE, blk, 0, stream>>>(xh, Wqh, bq, nullptr, Qh, M, EE, EE);
    gemm_bf16_kernel<EPI_ELU1><<<gE, blk, 0, stream>>>(xh, Wkh, bk, nullptr, Kh, M, EE, EE);
    gemm_bf16_kernel<EPI_BF16><<<gE, blk, 0, stream>>>(xh, Wvh, bv, nullptr, Vh, M, EE, EE);

    // KV / Ksum
    hipMemsetAsync(KV, 0, (size_t)BB * NHH * DHH * DHH * 4, stream);
    hipMemsetAsync(Ksum, 0, (size_t)BB * NHH * DHH * 4, stream);
    kv_ksum_kernel<<<dim3(32, BB * NHH), blk, 0, stream>>>(Kh, Vh, KV, Ksum);

    // attention combine
    attn_combine_kernel<<<dim3(SS / 32, BB * NHH), blk, 0, stream>>>(Qh, KV, Ksum, Ah);

    // output projection + LN1
    gemm_bf16_kernel<EPI_F32><<<gE, blk, 0, stream>>>(Ah, Woh, bo, t0, nullptr, M, EE, EE);
    add_ln_kernel<true><<<M, blk, 0, stream>>>(x, t0, g1, be1, x1, x1h);

    // FFN + LN2
    gemm_bf16_kernel<EPI_RELU><<<gH, blk, 0, stream>>>(x1h, W1h, b1, nullptr, Hh, M, HIDD, EE);
    gemm_bf16_kernel<EPI_F32><<<gE, blk, 0, stream>>>(Hh, W2h, b2, t0, nullptr, M, EE, HIDD);
    add_ln_kernel<false><<<M, blk, 0, stream>>>(x1, t0, g2, be2, (float*)d_out, nullptr);
}
